// ImpalaShadowNet_72258529787945
// MI455X (gfx1250) — compile-verified, matrix operand order checked by reading, not run on hardware
//
#include <hip/hip_runtime.h>
#include <hip/hip_bf16.h>

// ---------------------------------------------------------------------------
// IMPALA shadow net on MI455X (gfx1250), wave32 + WMMA bf16 path.
//   conv stack  -> implicit-GEMM v_wmma_f32_16x16x32_bf16, weights staged to
//                  LDS via GLOBAL_LOAD_ASYNC_TO_LDS (ASYNCcnt) when available
//   lstm x-proj -> blocked WMMA GEMM (2048x1024x1152), B panels async-staged
//   recurrence  -> single persistent 1024-thread block, h in LDS (bf16),
//                  c in registers, 32 waves x (2x2) 16x16 WMMA tiles/step
//   fc head     -> tiny f32 VALU kernel (18 cols, not worth a WMMA tile pad)
// ---------------------------------------------------------------------------

typedef __attribute__((ext_vector_type(16))) __bf16 v16bf;
typedef __attribute__((ext_vector_type(8)))  float  v8f;
typedef __attribute__((ext_vector_type(4)))  int    v4i_;

union ABFrag {
    v16bf          v;
    unsigned short u[16];
    uint4          q[2];
};

__device__ __forceinline__ unsigned short f32_to_bf16(float f) {
    union { float f; unsigned int u; } x; x.f = f;
    unsigned int u = x.u;
    unsigned int r = u + 0x7FFFu + ((u >> 16) & 1u);   // round-to-nearest-even
    return (unsigned short)(r >> 16);
}

__device__ __forceinline__ float sigf(float x) {
    return 1.0f / (1.0f + __expf(-x));
}

// ---- async global->LDS staging (CDNA5 ASYNCcnt path, with safe fallback) --
#if defined(__AMDGCN__) && __has_builtin(__builtin_amdgcn_global_load_async_to_lds_b128)
#define HAVE_ASYNC_LDS 1
#else
#define HAVE_ASYNC_LDS 0
#endif

__device__ __forceinline__ void async_copy_b128(const unsigned short* g,
                                                unsigned short* l) {
#if HAVE_ASYNC_LDS
    __builtin_amdgcn_global_load_async_to_lds_b128(
        (__attribute__((address_space(1))) v4i_*)g,
        (__attribute__((address_space(3))) v4i_*)l, 0, 0);
#else
    *(uint4*)l = *(const uint4*)g;
#endif
}

__device__ __forceinline__ void async_wait0() {
#if HAVE_ASYNC_LDS
#if __has_builtin(__builtin_amdgcn_s_wait_asynccnt)
    __builtin_amdgcn_s_wait_asynccnt(0);
#else
    asm volatile("s_wait_asynccnt 0x0" ::: "memory");
#endif
#endif
}

// K-index mapping for 16-bit A/B fragments (ISA 7.12.2, 16x32 bf16):
// lane half h (=lane>>4): element i<8  -> k = h*8 + i
//                         element i>=8 -> k = 16 + h*8 + (i-8)
__device__ __forceinline__ int frag_k(int half, int i) {
    return (i < 8) ? (half * 8 + i) : (16 + half * 8 + (i - 8));
}

// --------------------------- prep kernels ----------------------------------

__global__ void k_cast_bf16(const float* __restrict__ src,
                            unsigned short* __restrict__ dst, int n) {
    int i = blockIdx.x * blockDim.x + threadIdx.x;
    if (i < n) dst[i] = f32_to_bf16(src[i]);
}

// HWIO conv weights flatten to row-major [K, 32]; cast + zero-pad K to Kpad.
__global__ void k_prep_conv_w(const float* __restrict__ w,
                              unsigned short* __restrict__ out,
                              int Kreal, int Kpad) {
    int i = blockIdx.x * blockDim.x + threadIdx.x;
    if (i >= Kpad * 32) return;
    int k = i >> 5;
    out[i] = (k < Kreal) ? f32_to_bf16(w[i]) : (unsigned short)0;
}

// --------------------------- conv as implicit GEMM -------------------------
// Each wave: one 16-pixel x 16-channel output tile. N=32 -> 2 octiles.
// Zero-padded [Kpad,32] bf16 weights staged once per block into LDS via the
// async engine, B fragments then read from LDS.
__global__ void k_conv_wmma(const unsigned short* __restrict__ in,
                            const unsigned short* __restrict__ wts,
                            const float* __restrict__ bias,
                            unsigned short* __restrict__ out,
                            int Hin, int Win, int Hout, int Wout,
                            int log2Cin, int Kreal, int Ktiles, int pad,
                            int Mtiles) {
    __shared__ unsigned short wlds[288 * 32];      // max 18 KB (Kpad=288)

    int tid  = threadIdx.x;
    int wave = tid >> 5;
    int lane = tid & 31;

    // ---- stage weights (Kpad*32 bf16) into LDS ----
    int KW = Ktiles * 1024;                        // Kpad * 32 elems
    for (int idx = tid * 8; idx < KW; idx += blockDim.x * 8)
        async_copy_b128(wts + idx, &wlds[idx]);
    async_wait0();
    __syncthreads();

    int gw     = blockIdx.x * (blockDim.x >> 5) + wave;
    int ptile  = gw >> 1;
    int octile = gw & 1;
    if (ptile >= Mtiles) return;                   // wave-uniform exit

    int half = lane >> 4;
    int l16  = lane & 15;
    int Cin  = 1 << log2Cin;
    int cmask = Cin - 1;

    int pixel = ptile * 16 + l16;                  // A-fragment row
    int hw  = Hout * Wout;
    int img = pixel / hw;
    int rem = pixel - img * hw;
    int oy  = rem / Wout;
    int ox  = rem - oy * Wout;
    int iy0 = oy * 2 - pad;
    int ix0 = ox * 2 - pad;
    const unsigned short* imgbase = in + (size_t)img * Hin * Win * Cin;
    int col = octile * 16 + l16;                   // B/D column

    v8f acc = {};
    for (int kt = 0; kt < Ktiles; ++kt) {
        int k0 = kt * 32;
        ABFrag a, b;
#pragma unroll
        for (int i = 0; i < 16; ++i) {
            int kk = k0 + frag_k(half, i);
            unsigned short av = 0;
            if (kk < Kreal) {
                int kpos = kk >> log2Cin;          // 0..8  (ky*3+kx)
                int ci   = kk & cmask;
                int ky   = kpos / 3;
                int kx   = kpos - ky * 3;
                int iy = iy0 + ky, ix = ix0 + kx;
                if (iy >= 0 && iy < Hin && ix >= 0 && ix < Win)
                    av = imgbase[((size_t)iy * Win + ix) * Cin + ci];
            }
            a.u[i] = av;
            b.u[i] = wlds[kk * 32 + col];          // zero-padded beyond Kreal
        }
        acc = __builtin_amdgcn_wmma_f32_16x16x32_bf16(
            false, a.v, false, b.v, (short)0, acc, false, false);
    }

    float bcol = bias[col];
#pragma unroll
    for (int v = 0; v < 8; ++v) {
        int m = ptile * 16 + v + 8 * half;
        float f = acc[v] + bcol;
        f = (f > 0.0f) ? f : (__expf(f) - 1.0f);   // ELU
        out[(size_t)m * 32 + col] = f32_to_bf16(f);
    }
}

// --------------------------- blocked bf16 WMMA GEMM ------------------------
// C[M,N](f32) = A[M,K](bf16,rm) @ B[K,N](bf16,rm) + bias
// Block = 8 waves (2x4 tile grid) = 32x64 output tile. Per k-step the 32x64
// B panel is async-staged into LDS (one b128 per thread), then all waves
// build B fragments from LDS; A fragments are contiguous 16B global loads.
__global__ __launch_bounds__(256) void
k_gemm_bf16(const unsigned short* __restrict__ A,
            const unsigned short* __restrict__ B,
            const float* __restrict__ bias,
            float* __restrict__ C,
            int K, int ldn) {
    __shared__ unsigned short Bp[32 * 64];         // 4 KB panel

    int tid  = threadIdx.x;
    int wave = tid >> 5;
    int lane = tid & 31;
    int half = lane >> 4, l16 = lane & 15;

    int mt = blockIdx.y * 2 + (wave >> 2);
    int nt = blockIdx.x * 4 + (wave & 3);
    int nbase = blockIdx.x * 64;

    int arow = mt * 16 + l16;
    int col  = nt * 16 + l16;
    int colL = (wave & 3) * 16 + l16;              // column within LDS panel

    // staging source for this thread: row = tid>>3, col8 = (tid&7)*8
    int srow = tid >> 3;
    int scol = (tid & 7) * 8;

    v8f acc = {};
    for (int k0 = 0; k0 < K; k0 += 32) {
        async_copy_b128(B + (size_t)(k0 + srow) * ldn + nbase + scol,
                        &Bp[srow * 64 + scol]);
        async_wait0();
        __syncthreads();

        if (k0 + 32 < K) {                          // next A/B panels -> cache
            __builtin_prefetch(A + (size_t)arow * K + k0 + 32, 0, 0);
            __builtin_prefetch(B + (size_t)(k0 + 32 + srow) * ldn + nbase + scol, 0, 0);
        }

        ABFrag a, b;
        a.q[0] = *(const uint4*)(A + (size_t)arow * K + k0 + half * 8);
        a.q[1] = *(const uint4*)(A + (size_t)arow * K + k0 + 16 + half * 8);
#pragma unroll
        for (int i = 0; i < 16; ++i)
            b.u[i] = Bp[frag_k(half, i) * 64 + colL];
        acc = __builtin_amdgcn_wmma_f32_16x16x32_bf16(
            false, a.v, false, b.v, (short)0, acc, false, false);
        __syncthreads();                            // before panel overwrite
    }

    float bv = bias[col];
#pragma unroll
    for (int v = 0; v < 8; ++v) {
        int m = mt * 16 + v + 8 * half;
        C[(size_t)m * ldn + col] = acc[v] + bv;
    }
}

// --------------------------- persistent LSTM recurrence --------------------
#define NTJ  32
#define SZ   256
#define GSZ  1024
#define TLEN 64

__global__ __launch_bounds__(1024) void
k_lstm(const float* __restrict__ gx,              // [2048,1024], row = n*64+t
       const unsigned short* __restrict__ Wh,     // [256,1024] bf16
       const float* __restrict__ c0,
       const float* __restrict__ h0,
       float* __restrict__ hs) {                  // [2048,256],  row = n*64+t
    __shared__ unsigned short h_lds[NTJ][SZ];     //  16 KB
    __shared__ float          g_lds[NTJ][GSZ];    // 128 KB

    int tid  = threadIdx.x;
    int wave = tid >> 5, lane = tid & 31;
    int half = lane >> 4, l16 = lane & 15;

    for (int i = tid; i < NTJ * SZ; i += blockDim.x)
        h_lds[i >> 8][i & 255] = f32_to_bf16(h0[i]);

    float creg[8];                                // c state: 8 elems/thread
#pragma unroll
    for (int j = 0; j < 8; ++j) creg[j] = c0[tid * 8 + j];
    __syncthreads();

    int nt0 = wave * 2;                           // this wave's 2 N-tiles
    for (int t = 0; t < TLEN; ++t) {
        // ---- gates = gates_x[t] + h @ Wh  (WMMA) ----
        v8f acc[2][2];
#pragma unroll
        for (int mt = 0; mt < 2; ++mt)
#pragma unroll
            for (int ntl = 0; ntl < 2; ++ntl) {
                int col = (nt0 + ntl) * 16 + l16;
#pragma unroll
                for (int v = 0; v < 8; ++v) {
                    int n = mt * 16 + v + 8 * half;
                    acc[mt][ntl][v] = gx[((size_t)(n * TLEN + t)) * GSZ + col];
                }
            }
        for (int k0 = 0; k0 < SZ; k0 += 32) {
            ABFrag a[2], b[2];
#pragma unroll
            for (int mt = 0; mt < 2; ++mt) {
                int row = mt * 16 + l16;
                a[mt].q[0] = *(const uint4*)&h_lds[row][k0 + half * 8];
                a[mt].q[1] = *(const uint4*)&h_lds[row][k0 + 16 + half * 8];
            }
#pragma unroll
            for (int ntl = 0; ntl < 2; ++ntl) {
                int col = (nt0 + ntl) * 16 + l16;
#pragma unroll
                for (int i = 0; i < 16; ++i)
                    b[ntl].u[i] = Wh[(size_t)(k0 + frag_k(half, i)) * GSZ + col];
            }
#pragma unroll
            for (int mt = 0; mt < 2; ++mt)
#pragma unroll
                for (int ntl = 0; ntl < 2; ++ntl)
                    acc[mt][ntl] = __builtin_amdgcn_wmma_f32_16x16x32_bf16(
                        false, a[mt].v, false, b[ntl].v, (short)0,
                        acc[mt][ntl], false, false);
        }
#pragma unroll
        for (int mt = 0; mt < 2; ++mt)
#pragma unroll
            for (int ntl = 0; ntl < 2; ++ntl) {
                int col = (nt0 + ntl) * 16 + l16;
#pragma unroll
                for (int v = 0; v < 8; ++v)
                    g_lds[mt * 16 + v + 8 * half][col] = acc[mt][ntl][v];
            }
        __syncthreads();

        // ---- elementwise LSTM cell (i,j,f,o split, forget_bias=1) ----
#pragma unroll
        for (int j = 0; j < 8; ++j) {
            int flat = tid * 8 + j;
            int n = flat >> 8, col = flat & 255;
            float gi = g_lds[n][col];
            float gj = g_lds[n][col + 256];
            float gf = g_lds[n][col + 512];
            float go = g_lds[n][col + 768];
            float nc = creg[j] * sigf(gf + 1.0f) + sigf(gi) * tanhf(gj);
            float nh = tanhf(nc) * sigf(go);
            creg[j] = nc;
            h_lds[n][col] = f32_to_bf16(nh);
            hs[((size_t)(n * TLEN + t)) * SZ + col] = nh;
        }
        __syncthreads();
    }
}

// --------------------------- FC head ---------------------------------------
__global__ void k_fc(const float* __restrict__ feats,
                     const float* __restrict__ W,
                     const float* __restrict__ b,
                     float* __restrict__ out) {
    int idx = blockIdx.x * blockDim.x + threadIdx.x;
    if (idx >= 2048 * 18) return;
    int row = idx / 18, col = idx - row * 18;
    float s = b[col];
    for (int k = 0; k < 256; ++k)
        s += feats[(size_t)row * 256 + k] * W[k * 18 + col];
    out[idx] = s;
}

// ---------------------------------------------------------------------------
extern "C" void kernel_launch(void* const* d_in, const int* in_sizes, int n_in,
                              void* d_out, int out_size, void* d_ws,
                              size_t ws_size, hipStream_t stream) {
    (void)in_sizes; (void)n_in; (void)out_size; (void)ws_size;

    const float* inputs = (const float*)d_in[0];
    const float* w1 = (const float*)d_in[1];  const float* b1 = (const float*)d_in[2];
    const float* w2 = (const float*)d_in[3];  const float* b2 = (const float*)d_in[4];
    const float* w3 = (const float*)d_in[5];  const float* b3 = (const float*)d_in[6];
    const float* w4 = (const float*)d_in[7];  const float* b4 = (const float*)d_in[8];
    const float* lstm_w = (const float*)d_in[9];
    const float* lstm_b = (const float*)d_in[10];
    const float* fc_w   = (const float*)d_in[11];
    const float* fc_b   = (const float*)d_in[12];
    const float* c0     = (const float*)d_in[13];
    const float* h0     = (const float*)d_in[14];

    // ---- workspace layout (ping-pong bf16 activation buffers) ----
    const size_t IN_ELEMS = 2048ULL * 84 * 84 * 4;   // 57,802,752
    const size_t L1_ELEMS = 2048ULL * 42 * 42 * 32;  // 115,605,504
    size_t off = 0;
    auto take = [&](size_t bytes) {
        size_t r = off; off += (bytes + 255) & ~(size_t)255; return r;
    };
    unsigned short* buf0 = (unsigned short*)((char*)d_ws + take(IN_ELEMS * 2));
    unsigned short* buf1 = (unsigned short*)((char*)d_ws + take(L1_ELEMS * 2));
    unsigned short* wc1  = (unsigned short*)((char*)d_ws + take(64  * 32 * 2));
    unsigned short* wc2  = (unsigned short*)((char*)d_ws + take(288 * 32 * 2));
    unsigned short* wc3  = (unsigned short*)((char*)d_ws + take(288 * 32 * 2));
    unsigned short* wc4  = (unsigned short*)((char*)d_ws + take(288 * 32 * 2));
    unsigned short* wx   = (unsigned short*)((char*)d_ws + take(1152ULL * 1024 * 2));
    unsigned short* wh   = (unsigned short*)((char*)d_ws + take(256ULL * 1024 * 2));
    float* gx = (float*)((char*)d_ws + take(2048ULL * 1024 * 4));
    float* hs = (float*)((char*)d_ws + take(2048ULL * 256 * 4));

    // ---- casts / weight prep ----
    k_cast_bf16<<<(int)((IN_ELEMS + 255) / 256), 256, 0, stream>>>(
        inputs, buf0, (int)IN_ELEMS);
    k_prep_conv_w<<<8, 256, 0, stream>>>(w1, wc1, 36, 64);
    k_prep_conv_w<<<36, 256, 0, stream>>>(w2, wc2, 288, 288);
    k_prep_conv_w<<<36, 256, 0, stream>>>(w3, wc3, 288, 288);
    k_prep_conv_w<<<36, 256, 0, stream>>>(w4, wc4, 288, 288);
    k_cast_bf16<<<(1152 * 1024 + 255) / 256, 256, 0, stream>>>(
        lstm_w, wx, 1152 * 1024);
    k_cast_bf16<<<(256 * 1024 + 255) / 256, 256, 0, stream>>>(
        lstm_w + 1152 * 1024, wh, 256 * 1024);

    // ---- conv stack (implicit GEMM; block = 4 waves = 2 pixel-tiles) ----
    // L1: 84->42, Cin=4 (K 36->64), pad_before=0
    k_conv_wmma<<<112896, 128, 0, stream>>>(buf0, wc1, b1, buf1,
        84, 84, 42, 42, 2, 36, 2, 0, 225792);
    // L2: 42->21, Cin=32 (K=288), pad_before=0
    k_conv_wmma<<<28224, 128, 0, stream>>>(buf1, wc2, b2, buf0,
        42, 42, 21, 21, 5, 288, 9, 0, 56448);
    // L3: 21->11, pad_before=1
    k_conv_wmma<<<7744, 128, 0, stream>>>(buf0, wc3, b3, buf1,
        21, 21, 11, 11, 5, 288, 9, 1, 15488);
    // L4: 11->6, pad_before=1  -> Xfeat [2048,1152] bf16 in buf0
    k_conv_wmma<<<2304, 128, 0, stream>>>(buf1, wc4, b4, buf0,
        11, 11, 6, 6, 5, 288, 9, 1, 4608);

    // ---- gates_x = Xfeat @ Wx + lstm_b  (2048 x 1024 x 1152) ----
    // grid: 16 x-blocks * 4 ntiles = 64 N-tiles; 64 y-blocks * 2 = 128 M-tiles
    k_gemm_bf16<<<dim3(16, 64), 256, 0, stream>>>(buf0, wx, lstm_b, gx,
                                                  1152, 1024);

    // ---- 64-step recurrence, single persistent workgroup ----
    k_lstm<<<1, 1024, 0, stream>>>(gx, wh, c0, h0, hs);

    // ---- logits = hs @ fc_w + fc_b ----
    k_fc<<<288, 128, 0, stream>>>(hs, fc_w, fc_b, (float*)d_out);
}
